// Yolov8NMS_7121055777083
// MI455X (gfx1250) — matrix operand order repaired and checked by reading, not compiled, not run on hardware
//
#include <hip/hip_runtime.h>
#include <math.h>

typedef __attribute__((ext_vector_type(16))) _Float16 v16h;
typedef __attribute__((ext_vector_type(8)))  float    v8f;

#define NC        3
#define NM        32
#define N_ANCH    8400
#define IMG_S     640
#define MH        160
#define MW        160
#define MAX_DET   300
#define TOPK      1024
#define SORT_N    16384
#define MAX_WH    7680.0f
#define MASK_PIX  (IMG_S * IMG_S)     /* 409600 */
#define WORDS_PER (MASK_PIX / 32)     /* 12800  */
#define MTILES    19                  /* ceil(300/16) */
#define PROTO_HW  (MH * MW)           /* 25600 */

static __device__ __forceinline__ int iclamp(int v, int lo, int hi) {
  return v < lo ? lo : (v > hi ? hi : v);
}

// Fast sigmoid: 1-ulp v_rcp_f32 + fast exp. Only a 0.5 threshold (x==0
// crossing) ultimately matters downstream, so approximation error is benign.
static __device__ __forceinline__ float fast_sigmoid(float x) {
  return __builtin_amdgcn_rcpf(1.0f + __expf(-x));
}

// ---------------------------------------------------------------------------
// K0: per-anchor class argmax + sortable (score,idx) key. Pads to SORT_N.
// key = monotonic_u32(score) << 32 | (0xFFFFFFFF - idx)  (desc score, asc idx)
// ---------------------------------------------------------------------------
__global__ void k_decode(const float* __restrict__ pred,
                         const float* __restrict__ conf_thres,
                         unsigned long long* __restrict__ keys,
                         int* __restrict__ clsbuf) {
  int a = blockIdx.x * blockDim.x + threadIdx.x;
  if (a >= SORT_N) return;
  if (a >= N_ANCH) { keys[a] = 0ull; return; }
  float thr = conf_thres[0];
  float best = pred[4 * N_ANCH + a];
  int cls = 0;
#pragma unroll
  for (int c = 1; c < NC; ++c) {
    float v = pred[(4 + c) * N_ANCH + a];
    if (v > best) { best = v; cls = c; }
  }
  clsbuf[a] = cls;
  float score = (best > thr) ? best : -1.0f;
  unsigned int b = __float_as_uint(score);
  unsigned int mapped = (score >= 0.0f) ? (b | 0x80000000u) : ~b;
  keys[a] = ((unsigned long long)mapped << 32) |
            (unsigned long long)(0xFFFFFFFFu - (unsigned int)a);
}

// ---------------------------------------------------------------------------
// K1: single-workgroup bitonic sort (descending) over SORT_N u64 keys in
// global scratch, then emit top-1024 (score, anchor idx).
// ---------------------------------------------------------------------------
__global__ void k_sort_topk(unsigned long long* __restrict__ keys,
                            int* __restrict__ top_idx,
                            float* __restrict__ top_score) {
  const int tid = threadIdx.x;   // 1024 threads
  for (int k = 2; k <= SORT_N; k <<= 1) {
    for (int j = k >> 1; j > 0; j >>= 1) {
      for (int i = tid; i < SORT_N; i += TOPK) {
        int l = i ^ j;
        if (l > i) {
          unsigned long long x = keys[i], y = keys[l];
          bool desc = ((i & k) == 0);
          bool sw = desc ? (x < y) : (x > y);
          if (sw) { keys[i] = y; keys[l] = x; }
        }
      }
      __threadfence();
      __syncthreads();
    }
  }
  if (tid < TOPK) {
    unsigned long long key = keys[tid];
    unsigned int mapped = (unsigned int)(key >> 32);
    unsigned int fb = (mapped & 0x80000000u) ? (mapped & 0x7FFFFFFFu) : ~mapped;
    top_score[tid] = __uint_as_float(fb);
    top_idx[tid] = (int)(0xFFFFFFFFu - (unsigned int)(key & 0xFFFFFFFFull));
  }
}

// ---------------------------------------------------------------------------
// K2: greedy class-offset NMS over the top-1024 (reference fori_loop order).
// ---------------------------------------------------------------------------
__global__ void k_nms(const float* __restrict__ pred,
                      const int* __restrict__ clsbuf,
                      const int* __restrict__ top_idx,
                      const float* __restrict__ top_score,
                      const float* __restrict__ conf_thres,
                      const float* __restrict__ iou_thres,
                      int* __restrict__ nms_keep) {
  __shared__ float sx1[TOPK], sy1[TOPK], sx2[TOPK], sy2[TOPK], sar[TOPK];
  __shared__ unsigned char skeep[TOPK];
  const int t = threadIdx.x;   // 1024 threads
  float thr = conf_thres[0], ithr = iou_thres[0];
  int a = top_idx[t];
  float score = top_score[t];
  float cx = pred[0 * N_ANCH + a], cy = pred[1 * N_ANCH + a];
  float w  = pred[2 * N_ANCH + a], h  = pred[3 * N_ANCH + a];
  float off = (float)clsbuf[a] * MAX_WH;
  float x1 = cx - 0.5f * w + off, y1 = cy - 0.5f * h + off;
  float x2 = cx + 0.5f * w + off, y2 = cy + 0.5f * h + off;
  sx1[t] = x1; sy1[t] = y1; sx2[t] = x2; sy2[t] = y2;
  sar[t] = (x2 - x1) * (y2 - y1);
  skeep[t] = (score > thr) ? 1 : 0;
  __syncthreads();
  for (int i = 0; i < TOPK; ++i) {
    if (skeep[i] && t > i) {
      float lx = fmaxf(x1, sx1[i]), ly = fmaxf(y1, sy1[i]);
      float rx = fminf(x2, sx2[i]), ry = fminf(y2, sy2[i]);
      float iw = fmaxf(rx - lx, 0.0f), ih = fmaxf(ry - ly, 0.0f);
      float inter = iw * ih;
      float iou = inter / (sar[t] + sar[i] - inter + 1e-7f);
      if (iou > ithr) skeep[t] = 0;
    }
    __syncthreads();
  }
  nms_keep[t] = skeep[t];
}

// ---------------------------------------------------------------------------
// K3: compact surviving detections (already conf-sorted) into the first
// MAX_DET slots; scatter mask coefs straight into the CDNA5 WMMA A-fragment
// layout (16-bit A 16x32: lanes 0-15 K{0..7,16..23}, lanes 16-31 K{8..15,24..31}).
// ---------------------------------------------------------------------------
__global__ void k_compact(const float* __restrict__ pred,
                          const int* __restrict__ clsbuf,
                          const int* __restrict__ top_idx,
                          const float* __restrict__ top_score,
                          const int* __restrict__ nms_keep,
                          float* __restrict__ det_score,
                          int* __restrict__ det_cls,
                          int* __restrict__ det_valid,
                          float* __restrict__ det_box,
                          _Float16* __restrict__ A_frag) {
  __shared__ int scan[TOPK];
  __shared__ int total;
  const int t = threadIdx.x;   // 1024 threads
  int k = nms_keep[t];
  scan[t] = k;
  __syncthreads();
  for (int d = 1; d < TOPK; d <<= 1) {
    int add = (t >= d) ? scan[t - d] : 0;
    __syncthreads();
    scan[t] += add;
    __syncthreads();
  }
  int rank = scan[t] - k;               // exclusive prefix
  if (t == TOPK - 1) total = scan[TOPK - 1];
  __syncthreads();
  int nk = total < MAX_DET ? total : MAX_DET;

  for (int i = t; i < MTILES * 32 * 16; i += TOPK) A_frag[i] = (_Float16)0.0f;
  if (t >= nk && t < MAX_DET) {
    det_score[t] = -1.0f;
    det_cls[t] = 0;
    det_valid[t] = 0;
    det_box[4 * t + 0] = det_box[4 * t + 1] = 0.0f;
    det_box[4 * t + 2] = det_box[4 * t + 3] = 0.0f;
  }
  __syncthreads();

  if (k && rank < MAX_DET) {
    int a = top_idx[t];
    float cx = pred[0 * N_ANCH + a], cy = pred[1 * N_ANCH + a];
    float w  = pred[2 * N_ANCH + a], h  = pred[3 * N_ANCH + a];
    det_box[4 * rank + 0] = cx - 0.5f * w;
    det_box[4 * rank + 1] = cy - 0.5f * h;
    det_box[4 * rank + 2] = cx + 0.5f * w;
    det_box[4 * rank + 3] = cy + 0.5f * h;
    det_score[rank] = top_score[t];
    det_cls[rank] = clsbuf[a];
    det_valid[rank] = 1;
    int tt = rank >> 4, rr = rank & 15;
#pragma unroll
    for (int c = 0; c < NM; ++c) {
      float m = pred[(4 + NC + c) * N_ANCH + a];
      int lane = rr + 16 * ((c >> 3) & 1);
      int pos = (c & 7) + 8 * (c >> 4);
      A_frag[(tt * 32 + lane) * 16 + pos] = (_Float16)m;
    }
  }
}

// ---------------------------------------------------------------------------
// K4: pre-transpose proto (f32, [32][25600]) into per-lane f16 B-fragments:
// Bp[part][n][k'] = proto[(k'&8)*2+part*8+(k'&7)][n]  -> each WMMA lane does one 32B load.
// ---------------------------------------------------------------------------
__global__ void k_protopack(const float* __restrict__ proto,
                            _Float16* __restrict__ Bp) {
  int i = blockIdx.x * blockDim.x + threadIdx.x;   // 0 .. 2*25600-1
  if (i >= 2 * PROTO_HW) return;
  int part = i / PROTO_HW;
  int n = i - part * PROTO_HW;
  _Float16* dst = Bp + (size_t)i * 16;
#pragma unroll
  for (int kk = 0; kk < 16; ++kk)
    dst[kk] = (_Float16)proto[(size_t)((kk & 8) * 2 + part * 8 + (kk & 7)) * PROTO_HW + n];
}

// ---------------------------------------------------------------------------
// K5: mask GEMM via v_wmma_f32_16x16x32_f16, fused fast-sigmoid + box crop.
// 4 waves per 128-thread block; each wave owns one 16x16 tile (K=32 == 1 WMMA).
// ---------------------------------------------------------------------------
__global__ void __launch_bounds__(128)
k_maskgemm(const _Float16* __restrict__ A_frag,
           const _Float16* __restrict__ Bp,
           const float* __restrict__ det_box,
           _Float16* __restrict__ maskslo) {
  const int lane = threadIdx.x & 31;
  const int wave = threadIdx.x >> 5;            // 0..3
  const int mt = blockIdx.y;                    // 0..MTILES-1
  const int nt = blockIdx.x * 4 + wave;         // N-tile index
  const int n0 = nt * 16;                       // flattened HW column base

  v16h a = *(const v16h*)(A_frag + (size_t)(mt * 32 + lane) * 16);
  int part = lane >> 4;
  int col = n0 + (lane & 15);
  v16h b = *(const v16h*)(Bp + (size_t)(part * PROTO_HW + col) * 16);
  v8f c = {};
  c = __builtin_amdgcn_wmma_f32_16x16x32_f16(
      /*neg_a=*/false, a, /*neg_b=*/false, b,
      /*c_mod=*/(short)0, c, /*reuse_a=*/false, /*reuse_b=*/false);

  float accf[8];
  *(v8f*)accf = c;

  const int nb = n0 + (lane & 15);
  const int y = nb / MW;                        // same row for whole tile
  const int x = nb - y * MW;
  const float fx = (float)x, fy = (float)y;
#pragma unroll
  for (int v = 0; v < 8; ++v) {
    int m = mt * 16 + v + 8 * (lane >> 4);      // C/D layout: M = v + 8*(lane>=16)
    if (m < MAX_DET) {
      float bx1 = det_box[4 * m + 0] * 0.25f, by1 = det_box[4 * m + 1] * 0.25f;
      float bx2 = det_box[4 * m + 2] * 0.25f, by2 = det_box[4 * m + 3] * 0.25f;
      float s = fast_sigmoid(accf[v]);
      bool in = (fx >= bx1) && (fx < bx2) && (fy >= by1) && (fy < by2);
      maskslo[(size_t)m * PROTO_HW + nb] = (_Float16)(in ? s : 0.0f);
    }
  }
}

// ---------------------------------------------------------------------------
// K6: bilinear 160->640 upsample, threshold 0.5, det_valid gate, bit-pack
// 32 horizontal pixels per u32 with a wave32 ballot.
// ---------------------------------------------------------------------------
__global__ void k_resize_pack(const _Float16* __restrict__ maskslo,
                              const int* __restrict__ det_valid,
                              unsigned int* __restrict__ Mbits) {
  const int d = blockIdx.y;
  const int p = blockIdx.x * blockDim.x + threadIdx.x;   // 0..409599
  bool on = false;
  if (det_valid[d]) {
    int y = p / IMG_S, x = p - y * IMG_S;
    float fy = ((float)y + 0.5f) * 0.25f - 0.5f;
    float fx = ((float)x + 0.5f) * 0.25f - 0.5f;
    int y0f = (int)floorf(fy); float wy = fy - (float)y0f;
    int x0f = (int)floorf(fx); float wx = fx - (float)x0f;
    int y0 = iclamp(y0f, 0, MH - 1), y1 = iclamp(y0f + 1, 0, MH - 1);
    int x0 = iclamp(x0f, 0, MW - 1), x1 = iclamp(x0f + 1, 0, MW - 1);
    const _Float16* base = maskslo + (size_t)d * PROTO_HW;
    float v00 = (float)base[y0 * MW + x0], v01 = (float)base[y0 * MW + x1];
    float v10 = (float)base[y1 * MW + x0], v11 = (float)base[y1 * MW + x1];
    float v0 = v00 + (v01 - v00) * wx;
    float v1 = v10 + (v11 - v10) * wx;
    on = (v0 + (v1 - v0) * wy) > 0.5f;
  }
  unsigned int bal = __builtin_amdgcn_ballot_w32(on);
  if ((threadIdx.x & 31) == 0)
    Mbits[(size_t)d * WORDS_PER + (p >> 5)] = bal;
}

__global__ void k_zero_counts(float* __restrict__ counts) {
  if (threadIdx.x < 4) counts[threadIdx.x] = 0.0f;
}

// ---------------------------------------------------------------------------
// K7: einsum('ij,jhw->ihw') with P = "strictly higher conf & valid" on a
// conf-sorted list of binary masks == cumulative OR per pixel word:
// new_i = m_i & ~cum ; cum |= m_i(valid). Popcount -> per-class pixel counts.
// ---------------------------------------------------------------------------
__global__ void k_suppress(const unsigned int* __restrict__ Mbits,
                           const int* __restrict__ det_valid,
                           const int* __restrict__ det_cls,
                           unsigned int* __restrict__ Sbits,
                           float* __restrict__ counts) {
  int w = blockIdx.x * blockDim.x + threadIdx.x;
  if (w >= WORDS_PER) return;
  unsigned int cum = 0u;
  int cnt0 = 0, cnt1 = 0, cnt2 = 0;
  for (int i = 0; i < MAX_DET; ++i) {
    unsigned int s = Mbits[(size_t)i * WORDS_PER + w];
    unsigned int neu = s & ~cum;
    Sbits[(size_t)i * WORDS_PER + w] = neu;
    if (det_valid[i]) cum |= s;
    int pc = __popc(neu);
    int ci = det_cls[i];
    cnt0 += (ci == 0) ? pc : 0;
    cnt1 += (ci == 1) ? pc : 0;
    cnt2 += (ci == 2) ? pc : 0;
  }
  if (cnt0) atomicAdd(&counts[0], (float)cnt0);
  if (cnt1) atomicAdd(&counts[1], (float)cnt1);
  if (cnt2) atomicAdd(&counts[2], (float)cnt2);
}

__global__ void k_pxcount(const float* __restrict__ counts,
                          float* __restrict__ out) {
  if (threadIdx.x == 0 && blockIdx.x == 0) {
    float s = counts[0] + counts[1] + counts[2];
    out[0] = counts[0] / s;
    out[1] = counts[1] / s;
    out[2] = counts[2] / s;
  }
}

// ---------------------------------------------------------------------------
// K8: sequential alpha blend (reference scan order), CHW img -> HWC output.
// colors = [[0,1,1],[1,1,0],[0,1,0]], OPACITY=0.5.
// ---------------------------------------------------------------------------
__global__ void k_blend(const float* __restrict__ img,
                        const unsigned int* __restrict__ Sbits,
                        const int* __restrict__ det_cls,
                        float* __restrict__ out) {
  int p = blockIdx.x * blockDim.x + threadIdx.x;
  if (p >= MASK_PIX) return;
  float r = img[p];
  float g = img[MASK_PIX + p];
  float b = img[2 * MASK_PIX + p];
  const float colR[NC] = {0.0f, 1.0f, 0.0f};
  const float colG[NC] = {1.0f, 1.0f, 1.0f};
  const float colB[NC] = {1.0f, 0.0f, 0.0f};
  int word = p >> 5, bit = p & 31;
  for (int i = 0; i < MAX_DET; ++i) {
    if (i + 1 < MAX_DET)
      __builtin_prefetch(&Sbits[(size_t)(i + 1) * WORDS_PER + word], 0, 0);
    unsigned int wv = Sbits[(size_t)i * WORDS_PER + word];
    if ((wv >> bit) & 1u) {
      int ci = det_cls[i];
      r = 0.5f * r + 0.5f * colR[ci];
      g = 0.5f * g + 0.5f * colG[ci];
      b = 0.5f * b + 0.5f * colB[ci];
    }
  }
  out[3 + 3 * p + 0] = r;
  out[3 + 3 * p + 1] = g;
  out[3 + 3 * p + 2] = b;
}

// ---------------------------------------------------------------------------
extern "C" void kernel_launch(void* const* d_in, const int* in_sizes, int n_in,
                              void* d_out, int out_size, void* d_ws, size_t ws_size,
                              hipStream_t stream) {
  const float* pred       = (const float*)d_in[0];   // (1,39,8400)
  const float* proto      = (const float*)d_in[1];   // (32,160,160)
  const float* img        = (const float*)d_in[2];   // (1,3,640,640)
  const float* conf_thres = (const float*)d_in[3];
  const float* iou_thres  = (const float*)d_in[4];
  float* out = (float*)d_out;                        // [3 px_count | 640*640*3 HWC]

  char* ws = (char*)d_ws;
  size_t off = 0;
  auto bump = [&](size_t bytes) -> char* {
    char* p = ws + off;
    off += (bytes + 255) & ~(size_t)255;
    return p;
  };
  unsigned long long* keys = (unsigned long long*)bump(SORT_N * 8);
  int*   clsbuf    = (int*)bump(N_ANCH * 4);
  int*   top_idx   = (int*)bump(TOPK * 4);
  float* top_score = (float*)bump(TOPK * 4);
  int*   nms_keep  = (int*)bump(TOPK * 4);
  float* det_score = (float*)bump(MAX_DET * 4);
  int*   det_cls   = (int*)bump(MAX_DET * 4);
  int*   det_valid = (int*)bump(MAX_DET * 4);
  float* det_box   = (float*)bump(MAX_DET * 4 * 4);
  _Float16* A_frag = (_Float16*)bump((size_t)MTILES * 32 * 16 * 2);
  _Float16* Bp     = (_Float16*)bump((size_t)2 * PROTO_HW * 16 * 2);
  _Float16* maskslo= (_Float16*)bump((size_t)MAX_DET * PROTO_HW * 2);
  unsigned int* Mbits = (unsigned int*)bump((size_t)MAX_DET * WORDS_PER * 4);
  unsigned int* Sbits = (unsigned int*)bump((size_t)MAX_DET * WORDS_PER * 4);
  float* counts = (float*)bump(4 * 4);
  (void)in_sizes; (void)n_in; (void)out_size; (void)ws_size;

  k_decode<<<SORT_N / 256, 256, 0, stream>>>(pred, conf_thres, keys, clsbuf);
  k_sort_topk<<<1, TOPK, 0, stream>>>(keys, top_idx, top_score);
  k_nms<<<1, TOPK, 0, stream>>>(pred, clsbuf, top_idx, top_score,
                                conf_thres, iou_thres, nms_keep);
  k_compact<<<1, TOPK, 0, stream>>>(pred, clsbuf, top_idx, top_score, nms_keep,
                                    det_score, det_cls, det_valid, det_box, A_frag);
  k_protopack<<<(2 * PROTO_HW + 255) / 256, 256, 0, stream>>>(proto, Bp);
  k_maskgemm<<<dim3(PROTO_HW / 64, MTILES), 128, 0, stream>>>(A_frag, Bp, det_box, maskslo);
  k_resize_pack<<<dim3(MASK_PIX / 256, MAX_DET), 256, 0, stream>>>(maskslo, det_valid, Mbits);
  k_zero_counts<<<1, 32, 0, stream>>>(counts);
  k_suppress<<<(WORDS_PER + 255) / 256, 256, 0, stream>>>(Mbits, det_valid, det_cls,
                                                          Sbits, counts);
  k_pxcount<<<1, 32, 0, stream>>>(counts, out);
  k_blend<<<(MASK_PIX + 255) / 256, 256, 0, stream>>>(img, Sbits, det_cls, out);
}
